// ReluDecoder_20160576488055
// MI455X (gfx1250) — matrix operand order repaired and checked by reading, not compiled, not run on hardware
//
#include <hip/hip_runtime.h>
#include <hip/hip_bf16.h>

typedef __attribute__((ext_vector_type(16))) _Float16 v16h;
typedef __attribute__((ext_vector_type(8)))  _Float16 v8h;
typedef __attribute__((ext_vector_type(8)))  float    v8f;

#define Bn  16
#define Pn  512
#define Dn  20
#define En  256
#define NHn 4
#define Ln  12
#define Sn  1024
#define Mrows (Bn * Sn)   // 16384

#define WMMA_F16(a, b, c) \
  __builtin_amdgcn_wmma_f32_16x16x32_f16(false, (a), false, (b), (short)0, (c), false, false)

// A fragment (16x32, f16): lane L holds row M=L%16, K = kb..kb+7 and kb+16..kb+23, kb = (L>=16)?8:0
__device__ __forceinline__ v16h load_afrag(const _Float16* __restrict__ base, int ld,
                                           int m0, int k0, int lane) {
  int m  = m0 + (lane & 15);
  int kb = k0 + ((lane >> 4) << 3);
  const _Float16* p = base + (size_t)m * ld + kb;
  v8h lo = *(const v8h*)p;
  v8h hi = *(const v8h*)(p + 16);
  v16h r;
#pragma unroll
  for (int i = 0; i < 8; ++i) { r[i] = lo[i]; r[i + 8] = hi[i]; }
  return r;
}

// B fragment (32x16, f16) from BT[N][K]: lane L holds col N=L%16, K = kb..kb+7 and kb+16..kb+23, kb = (L>=16)?8:0
__device__ __forceinline__ v16h load_bfrag(const _Float16* __restrict__ bt, int ld,
                                           int n0, int k0, int lane) {
  int n  = n0 + (lane & 15);
  int kb = k0 + ((lane >> 4) << 3);
  const _Float16* p = bt + (size_t)n * ld + kb;
  v8h lo = *(const v8h*)p;
  v8h hi = *(const v8h*)(p + 16);
  v16h r;
#pragma unroll
  for (int i = 0; i < 8; ++i) { r[i] = lo[i]; r[i + 8] = hi[i]; }
  return r;
}

// -------- weight transpose + f32->f16 convert: WT[l][n][k] = W[l][k][n] --------
__global__ __launch_bounds__(256) void convT_kernel(const float* __restrict__ W,
                                                    _Float16* __restrict__ WT,
                                                    int K, int N, int total) {
  int idx = blockIdx.x * 256 + threadIdx.x;
  if (idx >= total) return;
  int l = idx / (N * K);
  int r = idx - l * (N * K);
  int n = r / K;
  int k = r - n * K;
  WT[idx] = (_Float16)W[(size_t)l * K * N + (size_t)k * N + n];
}

// -------- embed: H = combine(xs,ys) @ W_in + b_in --------
__global__ __launch_bounds__(256) void embed_kernel(const float* __restrict__ xs,
                                                    const float* __restrict__ ys,
                                                    const float* __restrict__ W_in,
                                                    const float* __restrict__ b_in,
                                                    float* __restrict__ H,
                                                    _Float16* __restrict__ H16) {
  int row = blockIdx.x;          // 0..16383
  int e   = threadIdx.x;         // 0..255
  int b = row >> 10;
  int s = row & (Sn - 1);
  int p = s >> 1;
  int odd = s & 1;
  const float* x = xs + ((size_t)b * Pn + p) * Dn;
  float acc = b_in[e];
#pragma unroll
  for (int d = 0; d < Dn; ++d) acc += x[d] * W_in[d * En + e];
  float last = odd ? ys[(size_t)b * Pn + p] : 0.0f;
  acc += last * W_in[Dn * En + e];
  H[(size_t)row * En + e]   = acc;
  H16[(size_t)row * En + e] = (_Float16)acc;
}

// -------- generic WMMA GEMM: C[MxN] = A[MxK] * BT[NxK]^T, block tile 128x64, wave tile 32x32
// mode 0: out f32 = C + bias
// mode 1: out f16 = relu(C + bias)
// mode 2: q/k: out f16 [B][NH][S][E] = C * head_mask[h]
// mode 3: v:   out f16 [B][NH][E][S] = C * head_mask[h]   (transposed store)
__global__ __launch_bounds__(256) void gemm_kernel(const _Float16* __restrict__ A,
                                                   const _Float16* __restrict__ BT,
                                                   int M, int N, int K,
                                                   const float* __restrict__ bias,
                                                   const float* __restrict__ head_mask,
                                                   float* __restrict__ outf,
                                                   _Float16* __restrict__ outh,
                                                   int mode) {
  int lane = threadIdx.x & 31;
  int wave = threadIdx.x >> 5;
  int wy = wave & 3, wx = wave >> 2;
  int mbase = blockIdx.x * 128 + wy * 32;
  int nbase = blockIdx.y * 64 + wx * 32;

  v8f acc[2][2] = {};
  for (int k0 = 0; k0 < K; k0 += 32) {
    v16h a0 = load_afrag(A, K, mbase,      k0, lane);
    v16h a1 = load_afrag(A, K, mbase + 16, k0, lane);
    v16h b0 = load_bfrag(BT, K, nbase,      k0, lane);
    v16h b1 = load_bfrag(BT, K, nbase + 16, k0, lane);
    acc[0][0] = WMMA_F16(a0, b0, acc[0][0]);
    acc[0][1] = WMMA_F16(a0, b1, acc[0][1]);
    acc[1][0] = WMMA_F16(a1, b0, acc[1][0]);
    acc[1][1] = WMMA_F16(a1, b1, acc[1][1]);
  }

  int rhi = (lane >> 4) << 3;
  int cl  = lane & 15;
#pragma unroll
  for (int ti = 0; ti < 2; ++ti)
#pragma unroll
    for (int tj = 0; tj < 2; ++tj) {
      int col = nbase + tj * 16 + cl;
#pragma unroll
      for (int r = 0; r < 8; ++r) {
        int row = mbase + ti * 16 + rhi + r;
        float v = acc[ti][tj][r];
        if (mode == 0) {
          outf[(size_t)row * N + col] = v + bias[col];
        } else if (mode == 1) {
          v += bias[col];
          v = v > 0.f ? v : 0.f;
          outh[(size_t)row * N + col] = (_Float16)v;
        } else {
          int b = row >> 10, s = row & (Sn - 1);
          int h = col >> 8,  e = col & (En - 1);
          v *= head_mask[h];
          if (mode == 2)
            outh[(((size_t)(b * NHn + h)) * Sn + s) * En + e] = (_Float16)v;
          else
            outh[(((size_t)(b * NHn + h)) * En + e) * Sn + s] = (_Float16)v;
        }
      }
    }
}

// -------- fused causal relu-attention, one wave per (b, h, 16-row i-block) --------
// ctxh[b,h,i,:] = sum_j ( relu(Q K^T) / (i+1) ) * V   (per-head partial; head sum in resln)
__global__ __launch_bounds__(128) void attn_kernel(const _Float16* __restrict__ Q,
                                                   const _Float16* __restrict__ Km,
                                                   const _Float16* __restrict__ VT,
                                                   float* __restrict__ ctxh) {
  __shared__ __align__(16) _Float16 sbuf[4][16 * 32];
  int lane = threadIdx.x & 31;
  int wave = threadIdx.x >> 5;
  int unit = blockIdx.x * 4 + wave;   // 0..4095 = ((b*NH + h)*64 + ib)
  int bh = unit >> 6;                  // b*NH + h
  int i0 = (unit & 63) << 4;
  _Float16* sm = sbuf[wave];

  v8f cacc[16] = {};
  int rhi = (lane >> 4) << 3;
  int cl  = lane & 15;
  int jend = i0 + 16;                 // causal: j blocks of 32 while j0 < i0+16

  const _Float16* Qh = Q  + ((size_t)bh * Sn + i0) * En;
  const _Float16* Kh = Km + (size_t)bh * Sn * En;
  const _Float16* Vh = VT + (size_t)bh * En * Sn;

  v16h qf[8];
#pragma unroll
  for (int t = 0; t < 8; ++t) qf[t] = load_afrag(Qh, En, 0, t * 32, lane);

  for (int j0 = 0; j0 < jend; j0 += 32) {
    // S tile: relu(Q K^T) * causal/(row+1), two 16x16 halves -> LDS (f16, row-major 16x32)
#pragma unroll
    for (int jt = 0; jt < 2; ++jt) {
      v8f s = {};
#pragma unroll
      for (int t = 0; t < 8; ++t) {
        v16h bf = load_bfrag(Kh, En, j0 + jt * 16, t * 32, lane);
        s = WMMA_F16(qf[t], bf, s);
      }
      int coln = j0 + jt * 16 + cl;
#pragma unroll
      for (int r = 0; r < 8; ++r) {
        int rown = i0 + rhi + r;
        float v = s[r];
        v = v > 0.f ? v : 0.f;
        v = (coln <= rown) ? v * (1.0f / (float)(rown + 1)) : 0.f;
        sm[(rhi + r) * 32 + jt * 16 + cl] = (_Float16)v;
      }
    }
    asm volatile("s_wait_dscnt 0" ::: "memory");   // intra-wave LDS RAW turnaround
    v16h af = load_afrag(sm, 32, 0, 0, lane);      // restriped attn tile as A-fragment
#pragma unroll
    for (int nt = 0; nt < 16; ++nt) {
      v16h bf = load_bfrag(Vh, Sn, nt * 16, j0, lane);  // V^T[e][j]
      cacc[nt] = WMMA_F16(af, bf, cacc[nt]);
    }
  }

  float* cp = ctxh + ((size_t)bh * Sn + i0) * En;
#pragma unroll
  for (int nt = 0; nt < 16; ++nt)
#pragma unroll
    for (int r = 0; r < 8; ++r)
      cp[(size_t)(rhi + r) * En + nt * 16 + cl] = cacc[nt][r];
}

// -------- residual add + layernorm (in-place on H), also emits f16 copy --------
// headsum=1: addv is per-head partial ctx [B][NH][S][E], sum the NH partials.
// headsum=0: addv is plain [row][E].
__global__ __launch_bounds__(256) void resln_kernel(const float* __restrict__ Hin,
                                                    const float* __restrict__ addv,
                                                    const float* __restrict__ g,
                                                    const float* __restrict__ beta,
                                                    float* __restrict__ Hout,
                                                    _Float16* __restrict__ H16,
                                                    int headsum) {
  __shared__ float red[8];
  int row = blockIdx.x, e = threadIdx.x;
  int lane = e & 31, wave = e >> 5;
  float x = Hin[(size_t)row * En + e];
  if (headsum) {
    int b = row >> 10, s = row & (Sn - 1);
#pragma unroll
    for (int h = 0; h < NHn; ++h)
      x += addv[(((size_t)(b * NHn + h)) * Sn + s) * En + e];
  } else {
    x += addv[(size_t)row * En + e];
  }

  float sum = x;
#pragma unroll
  for (int o = 16; o > 0; o >>= 1) sum += __shfl_xor(sum, o, 32);
  if (lane == 0) red[wave] = sum;
  __syncthreads();
  float mean = 0.f;
#pragma unroll
  for (int w = 0; w < 8; ++w) mean += red[w];
  mean *= (1.0f / En);
  __syncthreads();

  float d = x - mean;
  float vs = d * d;
#pragma unroll
  for (int o = 16; o > 0; o >>= 1) vs += __shfl_xor(vs, o, 32);
  if (lane == 0) red[wave] = vs;
  __syncthreads();
  float var = 0.f;
#pragma unroll
  for (int w = 0; w < 8; ++w) var += red[w];
  var *= (1.0f / En);

  float y = d * rsqrtf(var + 1e-5f) * g[e] + beta[e];
  Hout[(size_t)row * En + e] = y;
  H16[(size_t)row * En + e]  = (_Float16)y;
}

// -------- output head: out[b][p] = H[b][2p][:] . Wout + bout, one wave per output --------
__global__ __launch_bounds__(256) void out_kernel(const float* __restrict__ H,
                                                  const float* __restrict__ Wout,
                                                  const float* __restrict__ bout,
                                                  float* __restrict__ out) {
  int gid  = blockIdx.x * 256 + threadIdx.x;
  int wave = gid >> 5;
  int lane = threadIdx.x & 31;
  if (wave >= Bn * Pn) return;
  int b = wave >> 9;
  int p = wave & (Pn - 1);
  const float* h = H + ((size_t)(b * Sn + 2 * p)) * En;
  float acc = 0.f;
#pragma unroll
  for (int e = lane; e < En; e += 32) acc += h[e] * Wout[e];
#pragma unroll
  for (int o = 16; o > 0; o >>= 1) acc += __shfl_xor(acc, o, 32);
  if (lane == 0) out[(size_t)b * Pn + p] = acc + bout[0];
}

extern "C" void kernel_launch(void* const* d_in, const int* in_sizes, int n_in,
                              void* d_out, int out_size, void* d_ws, size_t ws_size,
                              hipStream_t stream) {
  const float* xs        = (const float*)d_in[0];
  const float* ys        = (const float*)d_in[1];
  const float* head_mask = (const float*)d_in[2];
  const float* W_in      = (const float*)d_in[3];
  const float* b_in      = (const float*)d_in[4];
  const float* Wq        = (const float*)d_in[5];
  const float* Wk        = (const float*)d_in[6];
  const float* Wv        = (const float*)d_in[7];
  const float* ln1_g     = (const float*)d_in[8];
  const float* ln1_b     = (const float*)d_in[9];
  const float* W1        = (const float*)d_in[10];
  const float* b1        = (const float*)d_in[11];
  const float* W2        = (const float*)d_in[12];
  const float* b2        = (const float*)d_in[13];
  const float* ln2_g     = (const float*)d_in[14];
  const float* ln2_b     = (const float*)d_in[15];
  const float* Wout      = (const float*)d_in[16];
  const float* bout      = (const float*)d_in[17];
  float* out             = (float*)d_out;

  char* base = (char*)d_ws;
  size_t off = 0;
  auto alloc = [&](size_t bytes) -> char* {
    char* p = base + off;
    off = (off + bytes + 255) & ~(size_t)255;
    return p;
  };

  float*    H    = (float*)   alloc((size_t)Mrows * En * 4);
  _Float16* H16  = (_Float16*)alloc((size_t)Mrows * En * 2);
  float*    CTX  = (float*)   alloc((size_t)Mrows * En * 4);             // mlp_out
  float*    CTXH = (float*)   alloc((size_t)Bn * NHn * Sn * En * 4);     // per-head ctx partials
  _Float16* Qb   = (_Float16*)alloc((size_t)Bn * NHn * Sn * En * 2);
  _Float16* Kb   = (_Float16*)alloc((size_t)Bn * NHn * Sn * En * 2);
  _Float16* VTb  = (_Float16*)alloc((size_t)Bn * NHn * En * Sn * 2);
  _Float16* ACT  = (_Float16*)alloc((size_t)Mrows * (4 * En) * 2);
  _Float16* WTq  = (_Float16*)alloc((size_t)Ln * (NHn * En) * En * 2);
  _Float16* WTk  = (_Float16*)alloc((size_t)Ln * (NHn * En) * En * 2);
  _Float16* WTv  = (_Float16*)alloc((size_t)Ln * (NHn * En) * En * 2);
  _Float16* WT1  = (_Float16*)alloc((size_t)Ln * (4 * En) * En * 2);
  _Float16* WT2  = (_Float16*)alloc((size_t)Ln * En * (4 * En) * 2);
  (void)ws_size; (void)in_sizes; (void)n_in; (void)out_size;

  // 1) weight transpose/convert (all layers at once per tensor)
  {
    int totQ = Ln * En * (NHn * En);                 // 12*256*1024
    dim3 g((totQ + 255) / 256);
    convT_kernel<<<g, 256, 0, stream>>>(Wq, WTq, En, NHn * En, totQ);
    convT_kernel<<<g, 256, 0, stream>>>(Wk, WTk, En, NHn * En, totQ);
    convT_kernel<<<g, 256, 0, stream>>>(Wv, WTv, En, NHn * En, totQ);
    convT_kernel<<<g, 256, 0, stream>>>(W1, WT1, En, 4 * En, totQ);
    convT_kernel<<<g, 256, 0, stream>>>(W2, WT2, 4 * En, En, totQ);
  }

  // 2) embed
  embed_kernel<<<dim3(Mrows), 256, 0, stream>>>(xs, ys, W_in, b_in, H, H16);

  // 3) layers
  const size_t wqkvStride = (size_t)(NHn * En) * En;  // 1024*256 per layer
  dim3 gQKV(Mrows / 128, (NHn * En) / 64);            // (128, 16)
  dim3 gMLP1(Mrows / 128, (4 * En) / 64);             // (128, 16)
  dim3 gMLP2(Mrows / 128, En / 64);                   // (128, 4)
  dim3 gATTN((Bn * NHn * (Sn / 16)) / 4);             // 4096 wave-units / 4 waves per block
  for (int l = 0; l < Ln; ++l) {
    gemm_kernel<<<gQKV, 256, 0, stream>>>(H16, WTq + (size_t)l * wqkvStride,
                                          Mrows, NHn * En, En, nullptr, head_mask,
                                          nullptr, Qb, 2);
    gemm_kernel<<<gQKV, 256, 0, stream>>>(H16, WTk + (size_t)l * wqkvStride,
                                          Mrows, NHn * En, En, nullptr, head_mask,
                                          nullptr, Kb, 2);
    gemm_kernel<<<gQKV, 256, 0, stream>>>(H16, WTv + (size_t)l * wqkvStride,
                                          Mrows, NHn * En, En, nullptr, head_mask,
                                          nullptr, VTb, 3);
    attn_kernel<<<gATTN, 128, 0, stream>>>(Qb, Kb, VTb, CTXH);
    resln_kernel<<<dim3(Mrows), 256, 0, stream>>>(H, CTXH, ln1_g + (size_t)l * En,
                                                  ln1_b + (size_t)l * En, H, H16, 1);
    gemm_kernel<<<gMLP1, 256, 0, stream>>>(H16, WT1 + (size_t)l * wqkvStride,
                                           Mrows, 4 * En, En, b1 + (size_t)l * 4 * En,
                                           nullptr, nullptr, ACT, 1);
    gemm_kernel<<<gMLP2, 256, 0, stream>>>(ACT, WT2 + (size_t)l * wqkvStride,
                                           Mrows, En, 4 * En, b2 + (size_t)l * En,
                                           nullptr, CTX, nullptr, 0);
    resln_kernel<<<dim3(Mrows), 256, 0, stream>>>(H, CTX, ln2_g + (size_t)l * En,
                                                  ln2_b + (size_t)l * En, H, H16, 0);
  }

  // 4) output head
  out_kernel<<<dim3((Bn * Pn * 32 + 255) / 256), 256, 0, stream>>>(H, Wout, bout, out);
}
